// Mask2FormerCriterion_70738111365368
// MI455X (gfx1250) — matrix operand order repaired and checked by reading, not compiled, not run on hardware
//
#include <hip/hip_runtime.h>
#include <hip/hip_bf16.h>

#define B_   4
#define Q_   100
#define C1_  20      // NUM_CLASSES + 1
#define M_   16
#define HW_  65536
#define P_   12544   // divisible by 32: 392 chunks
#define NCLS 19

#define LOG2E_F 1.4426950408889634f
#define LN2_F   0.6931471805599453f

typedef __attribute__((ext_vector_type(16))) _Float16 v16h;
typedef __attribute__((ext_vector_type(8)))  float    v8f;

// ---------------- threefry-2x32 (jax-style counter PRNG) ----------------
__device__ __forceinline__ unsigned rotl32(unsigned x, int r) {
    return (x << r) | (x >> (32 - r));
}

__device__ __forceinline__ void tf2x32(unsigned k0, unsigned k1,
                                       unsigned x0, unsigned x1,
                                       unsigned& o0, unsigned& o1) {
    const unsigned k2 = k0 ^ k1 ^ 0x1BD11BDAu;
    unsigned v0 = x0 + k0, v1 = x1 + k1;
#define TF_QR(r) { v0 += v1; v1 = rotl32(v1, r); v1 ^= v0; }
#define TF_4A TF_QR(13) TF_QR(15) TF_QR(26) TF_QR(6)
#define TF_4B TF_QR(17) TF_QR(29) TF_QR(16) TF_QR(24)
    TF_4A; v0 += k1; v1 += k2 + 1u;
    TF_4B; v0 += k2; v1 += k0 + 2u;
    TF_4A; v0 += k0; v1 += k1 + 3u;
    TF_4B; v0 += k1; v1 += k2 + 4u;
    TF_4A; v0 += k2; v1 += k0 + 5u;
#undef TF_4B
#undef TF_4A
#undef TF_QR
    o0 = v0; o1 = v1;
}

__device__ __forceinline__ unsigned tf_bits(unsigned k0, unsigned k1, unsigned ctr) {
    unsigned a, b_;
    tf2x32(k0, k1, 0u, ctr, a, b_);
    return a;
}

// ---- shared-exp sigmoid + softplus on raw HW transcendentals -----------
// exp arg <= 0 -> e in (0,1]; log arg in (1,2]: no denorm/inf/nan guards needed.
// v_exp_f32 (exp2) + v_rcp_f32 + v_log_f32 (log2) only.
__device__ __forceinline__ void sig_sp(float x, float& s, float& sp) {
    const float e = __builtin_amdgcn_exp2f(-fabsf(x) * LOG2E_F);   // v_exp_f32
    const float r = __builtin_amdgcn_rcpf(1.f + e);                // v_rcp_f32
    s  = (x >= 0.f) ? r : e * r;
    sp = fmaxf(x, 0.f) + __builtin_amdgcn_logf(1.f + e) * LN2_F;   // v_log_f32
}

// ---------------- kernel 0: zero the accumulators ----------------
__global__ void init_kernel(float* acc) {
    if (threadIdx.x < 4) acc[threadIdx.x] = 0.f;
}

// ---------------- kernel 1: per-batch matching (WMMA costs + Hungarian) --
// grid = B_ blocks x 256 threads (8 waves). Waves 0-6: 16-query WMMA tiles.
// Wave 7: sampled tf column sums. Thread 0: Jonker-Volgenant on 16x100.
__global__ __launch_bounds__(256)
void match_kernel(const float* __restrict__ pred_logits,
                  const float* __restrict__ pred_masks,
                  const float* __restrict__ tgt_masks,
                  const int*   __restrict__ tgt_labels,
                  int* __restrict__ pred_idx) {
    __shared__ float s_probs[Q_][C1_];     // softmax(pred_logits[b])
    __shared__ float s_cost[M_][112];      // cost^T: (target, query), padded
    __shared__ float s_tfsum[M_];
    __shared__ int   s_labels[M_];
    // Hungarian scratch (thread 0 only)
    __shared__ float h_u[M_ + 1], h_v[Q_ + 1], h_minv[Q_ + 1];
    __shared__ int   h_p[Q_ + 1], h_way[Q_ + 1];
    __shared__ unsigned char h_used[Q_ + 1];

    const int b    = blockIdx.x;
    const int tid  = threadIdx.x;
    const int lane = tid & 31;
    const int w    = tid >> 5;

    if (tid < M_) s_labels[tid] = tgt_labels[b * M_ + tid];
    if (tid < Q_) {
        const float* row = pred_logits + (size_t)(b * Q_ + tid) * C1_;
        float mx = row[0];
        #pragma unroll
        for (int c = 1; c < C1_; ++c) mx = fmaxf(mx, row[c]);
        float se = 0.f;
        #pragma unroll
        for (int c = 0; c < C1_; ++c) se += __builtin_amdgcn_exp2f((row[c] - mx) * LOG2E_F);
        const float inv = __builtin_amdgcn_rcpf(se);
        #pragma unroll
        for (int c = 0; c < C1_; ++c)
            s_probs[tid][c] = __builtin_amdgcn_exp2f((row[c] - mx) * LOG2E_F) * inv;
    }

    // folded key for this batch: fold_in(key(42), b)
    unsigned fk0, fk1;
    tf2x32(0u, 42u, 0u, (unsigned)b, fk0, fk1);

    v8f accd = {0.f, 0.f, 0.f, 0.f, 0.f, 0.f, 0.f, 0.f};  // pf @ tf^T
    v8f accs = {0.f, 0.f, 0.f, 0.f, 0.f, 0.f, 0.f, 0.f};  // sigmoid(pf) @ tf^T
    float sp = 0.f, sg = 0.f;                              // per-row softplus / sigmoid sums
    const int hi8 = (lane & 16) >> 1;                      // 0 or 8 (upper half-wave K shift)

    if (w < 7) {
        const int q0  = w * 16;
        const int qrc = min(q0 + (lane & 15), Q_ - 1);     // clamp padded rows
        const float* prow  = pred_masks + (size_t)(b * Q_ + qrc) * HW_;
        const float* tbase = tgt_masks  + (size_t)(b * M_) * HW_;

        for (int kb = 0; kb < P_; kb += 32) {
            // shared sampled point for this lane's K slot (K = lane)
            const unsigned pl = tf_bits(fk0, fk1, (unsigned)(kb + lane)) & (HW_ - 1u);

            // B tile 32x16 f16: column N = lane&15, K(e) = e + (e&8) + hi8
            v16h bv;
            #pragma unroll
            for (int e = 0; e < 16; ++e)
                bv[e] = (_Float16)tbase[(size_t)(lane & 15) * HW_ + (unsigned)__shfl((int)pl, e + (e & 8) + hi8, 32)];

            // A tiles 16x32 f16: row m = lane&15, K(e) = e + (e&8) + hi8
            v16h av, avs;
            #pragma unroll
            for (int e = 0; e < 16; ++e) {
                const int ke = e + (e & 8) + hi8;
                const unsigned idx = (unsigned)__shfl((int)pl, ke, 32);
                const float x = prow[idx];
                float s, spv;
                sig_sp(x, s, spv);
                av[e]  = (_Float16)x;
                avs[e] = (_Float16)s;
                sp += spv;
                sg += s;
            }
            accd = __builtin_amdgcn_wmma_f32_16x16x32_f16(false, av,  false, bv,
                                                          (short)0, accd, false, false);
            accs = __builtin_amdgcn_wmma_f32_16x16x32_f16(false, avs, false, bv,
                                                          (short)0, accs, false, false);
        }
        // combine half-wave partial row sums (lanes m and m+16 share row m)
        sp += __shfl_xor(sp, 16, 32);
        sg += __shfl_xor(sg, 16, 32);
    } else {
        // wave 7: sampled tf column sums over the same point set
        float ts[M_];
        #pragma unroll
        for (int n = 0; n < M_; ++n) ts[n] = 0.f;
        const float* tbase = tgt_masks + (size_t)(b * M_) * HW_;
        for (int p = lane; p < P_; p += 32) {
            const unsigned idx = tf_bits(fk0, fk1, (unsigned)p) & (HW_ - 1u);
            #pragma unroll
            for (int n = 0; n < M_; ++n) ts[n] += tbase[(size_t)n * HW_ + idx];
        }
        #pragma unroll
        for (int n = 0; n < M_; ++n) {
            float v = ts[n];
            for (int off = 16; off > 0; off >>= 1) v += __shfl_down(v, off, 32);
            if (lane == 0) s_tfsum[n] = v;
        }
    }
    __syncthreads();

    // assemble cost^T(target, query) from WMMA accumulators
    if (w < 7) {
        const int   n   = lane & 15;
        const float tfs = s_tfsum[n];
        const int   lbl = s_labels[n];
        const float invP = 1.f / (float)P_;
        #pragma unroll
        for (int r = 0; r < 8; ++r) {
            const int mloc = r + hi8;                 // C/D layout: M = r (+8 upper half)
            const int q    = w * 16 + mloc;
            const float sprow = __shfl(sp, mloc, 32);
            const float sgrow = __shfl(sg, mloc, 32);
            if (q < Q_) {
                const float cclass = -s_probs[q][lbl];
                const float cmask  = (sprow - accd[r]) * invP;
                const float cdice  = 1.f - (2.f * accs[r] + 1.f) *
                                     __builtin_amdgcn_rcpf(sgrow + tfs + 1.f);
                s_cost[n][q] = 2.f * cclass + 5.f * cmask + 5.f * cdice;
            }
        }
    }
    __syncthreads();

    // Jonker-Volgenant Hungarian, n = M_ rows, m = Q_ cols (sequential)
    if (tid == 0) {
        const float INF = 1e30f;
        for (int i = 0; i <= M_; ++i) h_u[i] = 0.f;
        for (int j = 0; j <= Q_; ++j) { h_v[j] = 0.f; h_p[j] = 0; h_way[j] = 0; }
        for (int i = 1; i <= M_; ++i) {
            h_p[0] = i;
            int j0 = 0;
            for (int j = 0; j <= Q_; ++j) { h_minv[j] = INF; h_used[j] = 0; }
            for (;;) {
                h_used[j0] = 1;
                const int i0 = h_p[j0];
                float delta = INF;
                int j1 = 0;
                for (int j = 1; j <= Q_; ++j) {
                    if (!h_used[j]) {
                        const float cur = s_cost[i0 - 1][j - 1] - h_u[i0] - h_v[j];
                        if (cur < h_minv[j]) { h_minv[j] = cur; h_way[j] = j0; }
                        if (h_minv[j] < delta) { delta = h_minv[j]; j1 = j; }
                    }
                }
                for (int j = 0; j <= Q_; ++j) {
                    if (h_used[j]) { h_u[h_p[j]] += delta; h_v[j] -= delta; }
                    else           { h_minv[j] -= delta; }
                }
                j0 = j1;
                if (h_p[j0] == 0) break;
            }
            while (j0 != 0) { const int j1 = h_way[j0]; h_p[j0] = h_p[j1]; j0 = j1; }
        }
        for (int j = 1; j <= Q_; ++j)
            if (h_p[j] > 0) pred_idx[b * M_ + (h_p[j] - 1)] = j - 1;
    }
}

// ---------------- kernel 2: weighted class cross-entropy ----------------
__global__ void ce_kernel(const float* __restrict__ pred_logits,
                          const int*   __restrict__ tgt_labels,
                          const int*   __restrict__ pred_idx,
                          float* __restrict__ acc) {
    const int i = blockIdx.x * blockDim.x + threadIdx.x;
    if (i >= B_ * Q_) return;
    const int b = i / Q_, q = i % Q_;
    const float* row = pred_logits + (size_t)i * C1_;
    float mx = row[0];
    #pragma unroll
    for (int c = 1; c < C1_; ++c) mx = fmaxf(mx, row[c]);
    float se = 0.f;
    #pragma unroll
    for (int c = 0; c < C1_; ++c) se += __builtin_amdgcn_exp2f((row[c] - mx) * LOG2E_F);
    const float lse = __builtin_amdgcn_logf(se) * LN2_F + mx;
    int cls = NCLS;
    #pragma unroll
    for (int m = 0; m < M_; ++m)
        if (pred_idx[b * M_ + m] == q) cls = tgt_labels[b * M_ + m];
    const float wgt = (cls == NCLS) ? 0.1f : 1.f;
    atomicAdd(&acc[0], wgt * (lse - row[cls]));
    atomicAdd(&acc[1], wgt);
}

// ---------------- kernel 3: point-sampled mask BCE + dice ---------------
// one block per matched pair (64 blocks x 256 threads)
__global__ __launch_bounds__(256)
void maskdice_kernel(const float* __restrict__ pred_masks,
                     const float* __restrict__ tgt_masks,
                     const int*   __restrict__ pred_idx,
                     float* __restrict__ acc) {
    const int n = blockIdx.x;            // 0..63
    const int b = n >> 4, m = n & 15;
    int q = pred_idx[b * M_ + m];
    q = min(max(q, 0), Q_ - 1);
    const float* prow = pred_masks + (size_t)(b * Q_ + q) * HW_;
    const float* trow = tgt_masks  + (size_t)(b * M_ + m) * HW_;

    unsigned fk0, fk1;
    tf2x32(0u, 42u, 0u, 10000u, fk0, fk1);   // fold_in(key(42), 10000)

    float bce = 0.f, num = 0.f, ss = 0.f, ts = 0.f;
    for (int p = threadIdx.x; p < P_; p += 256) {
        const unsigned idx = tf_bits(fk0, fk1, (unsigned)(n * P_ + p)) & (HW_ - 1u);
        const float x = prow[idx];
        const float t = trow[idx];
        float s, spv;
        sig_sp(x, s, spv);
        bce += spv - x * t;
        num += s * t; ss += s; ts += t;
    }
    const int lane = threadIdx.x & 31, w = threadIdx.x >> 5;
    for (int off = 16; off > 0; off >>= 1) {
        bce += __shfl_down(bce, off, 32);
        num += __shfl_down(num, off, 32);
        ss  += __shfl_down(ss,  off, 32);
        ts  += __shfl_down(ts,  off, 32);
    }
    __shared__ float red[8][4];
    if (lane == 0) { red[w][0] = bce; red[w][1] = num; red[w][2] = ss; red[w][3] = ts; }
    __syncthreads();
    if (threadIdx.x == 0) {
        float tb = 0.f, tn = 0.f, tss = 0.f, tts = 0.f;
        #pragma unroll
        for (int i = 0; i < 8; ++i) {
            tb += red[i][0]; tn += red[i][1]; tss += red[i][2]; tts += red[i][3];
        }
        atomicAdd(&acc[2], tb / (float)P_);
        atomicAdd(&acc[3], 1.f - (2.f * tn + 1.f) / (tss + tts + 1.f));
    }
}

// ---------------- kernel 4: finalize the 4 scalars ----------------------
__global__ void final_kernel(const float* __restrict__ acc, float* __restrict__ out) {
    const float ce = acc[0] / acc[1];
    const float lm = acc[2] / ((float)(B_ * M_) + 1e-8f);
    const float ld = acc[3] / ((float)(B_ * M_) + 1e-8f);
    out[0] = ce;
    out[1] = lm;
    out[2] = ld;
    out[3] = 2.f * ce + 5.f * lm + 5.f * ld;
}

extern "C" void kernel_launch(void* const* d_in, const int* in_sizes, int n_in,
                              void* d_out, int out_size, void* d_ws, size_t ws_size,
                              hipStream_t stream) {
    const float* pred_logits = (const float*)d_in[0];
    const float* pred_masks  = (const float*)d_in[1];
    const float* tgt_masks   = (const float*)d_in[2];
    const int*   tgt_labels  = (const int*)d_in[3];

    int*   pred_idx = (int*)d_ws;                                // 64 ints
    float* acc      = (float*)((char*)d_ws + 64 * sizeof(int));  // 4 floats
    float* out      = (float*)d_out;

    init_kernel<<<1, 64, 0, stream>>>(acc);
    match_kernel<<<B_, 256, 0, stream>>>(pred_logits, pred_masks, tgt_masks,
                                         tgt_labels, pred_idx);
    ce_kernel<<<(B_ * Q_ + 255) / 256, 256, 0, stream>>>(pred_logits, tgt_labels,
                                                         pred_idx, acc);
    maskdice_kernel<<<B_ * M_, 256, 0, stream>>>(pred_masks, tgt_masks, pred_idx, acc);
    final_kernel<<<1, 1, 0, stream>>>(acc, out);
}
